// GPT_14740327760420
// MI455X (gfx1250) — matrix operand order repaired and checked by reading, not compiled, not run on hardware
//
#include <hip/hip_runtime.h>
#include <hip/hip_bf16.h>
#include <math.h>

// ---------------- problem constants ----------------
constexpr int Bc   = 4;
constexpr int Sc   = 1024;
constexpr int Dc   = 512;
constexpr int Hc   = 8;
constexpr int Lc   = 4;
constexpr int DFFc = 2048;
constexpr int Vc   = 32000;
constexpr int BSc  = Bc * Sc;       // 4096 token rows
constexpr int HDc  = Hc * Dc;       // 4096

// ---------------- types ----------------
typedef __attribute__((ext_vector_type(16))) __bf16 v16bf;
typedef __attribute__((ext_vector_type(8)))  __bf16 v8bf;
typedef __attribute__((ext_vector_type(8)))  float  v8f;

__device__ __forceinline__ v16bf cat16(v8bf lo, v8bf hi) {
  v16bf r;
#pragma unroll
  for (int i = 0; i < 8; ++i) { r[i] = lo[i]; r[i + 8] = hi[i]; }
  return r;
}

// ---------------- WMMA GEMM ----------------
// C[m,n] = act(alpha * sum_k A[m,k] * B(k,n)) (+ R[m,n])
// BT=false: B is [K,N] row-major (ldb = row stride)
// BT=true : B is [N,K] row-major (C = A * B^T)
// Batched over blockIdx.z with z0 = z % Z0, z1 = z / Z0 and per-operand strides.
// 64x64x32 block tile, 4 waves (wave32), 32x32 per wave (4 WMMAs / K-step),
// double-buffered LDS so global loads overlap the matrix math.
constexpr int BM = 64, BN = 64, BK = 32, BKP = 56; // 112B row stride: 16B aligned

template<bool BT>
__global__ __launch_bounds__(128)
void gemm_wmma(const float* __restrict__ A, const float* __restrict__ Bm,
               const float* __restrict__ R, float* __restrict__ C,
               int M, int N, int K, int lda, int ldb, int ldc,
               int Z0,
               long long sA0, long long sA1,
               long long sB0, long long sB1,
               long long sC0, long long sC1,
               float alpha, int act)
{
  __shared__ __attribute__((aligned(16))) __bf16 sA[2][BM][BKP];
  __shared__ __attribute__((aligned(16))) __bf16 sB[2][BN][BKP];   // sB[.][n][k]

  const int tid  = threadIdx.x;
  const int lane = tid & 31;
  const int wave = tid >> 5;
  const int wm = (wave >> 1) * 32;
  const int wn = (wave & 1) * 32;
  const int m0 = blockIdx.y * BM;
  const int n0 = blockIdx.x * BN;
  const int z  = blockIdx.z;
  const int z0 = z % Z0;
  const int z1 = z / Z0;

  const float* Ab = A + z0 * sA0 + z1 * sA1 + (long long)m0 * lda;
  const float* Bb = BT ? (Bm + z0 * sB0 + z1 * sB1 + (long long)n0 * ldb)
                       : (Bm + z0 * sB0 + z1 * sB1 + n0);
  float*       Cb = C + z0 * sC0 + z1 * sC1;
  const float* Rb = R ? (R + z0 * sC0 + z1 * sC1) : nullptr;

  v8f acc[2][2] = {};

  float4 ra[4], rb[4];   // register staging for one (A,B) tile pair

  // Issue all global loads for tile @kk back-to-back (single wait at first use).
  auto load_tiles = [&](int kk) {
#pragma unroll
    for (int p = 0; p < 4; ++p) {
      int idx = tid + p * 128;
      int r = idx >> 3, c = (idx & 7) << 2;
      ra[p] = *(const float4*)(Ab + (long long)r * lda + kk + c);
    }
    if (BT) {
#pragma unroll
      for (int p = 0; p < 4; ++p) {
        int idx = tid + p * 128;
        int r = idx >> 3, c = (idx & 7) << 2;          // r = n, c = k
        rb[p] = *(const float4*)(Bb + (long long)r * ldb + kk + c);
      }
    } else {
#pragma unroll
      for (int p = 0; p < 4; ++p) {
        int idx = tid + p * 128;
        int kr = idx >> 4, c = (idx & 15) << 2;        // kr = k, c = n
        rb[p] = *(const float4*)(Bb + (long long)(kk + kr) * ldb + c);
      }
    }
  };

  // Convert f32 -> bf16 (native cvt) and store staged tile into LDS buffer.
  auto store_tiles = [&](int buf) {
#pragma unroll
    for (int p = 0; p < 4; ++p) {
      int idx = tid + p * 128;
      int r = idx >> 3, c = (idx & 7) << 2;
      sA[buf][r][c + 0] = (__bf16)ra[p].x; sA[buf][r][c + 1] = (__bf16)ra[p].y;
      sA[buf][r][c + 2] = (__bf16)ra[p].z; sA[buf][r][c + 3] = (__bf16)ra[p].w;
    }
    if (BT) {
#pragma unroll
      for (int p = 0; p < 4; ++p) {
        int idx = tid + p * 128;
        int r = idx >> 3, c = (idx & 7) << 2;
        sB[buf][r][c + 0] = (__bf16)rb[p].x; sB[buf][r][c + 1] = (__bf16)rb[p].y;
        sB[buf][r][c + 2] = (__bf16)rb[p].z; sB[buf][r][c + 3] = (__bf16)rb[p].w;
      }
    } else {
#pragma unroll
      for (int p = 0; p < 4; ++p) {
        int idx = tid + p * 128;
        int kr = idx >> 4, c = (idx & 15) << 2;
        sB[buf][c + 0][kr] = (__bf16)rb[p].x; sB[buf][c + 1][kr] = (__bf16)rb[p].y;
        sB[buf][c + 2][kr] = (__bf16)rb[p].z; sB[buf][c + 3][kr] = (__bf16)rb[p].w;
      }
    }
  };

  // prologue: stage tile 0
  load_tiles(0);
  store_tiles(0);
  int buf = 0;

  for (int kk = 0; kk < K; kk += BK) {
    const bool more = (kk + BK) < K;
    // Kick off next tile's global loads before the barrier (overlaps WMMAs).
    if (more) load_tiles(kk + BK);
    // Hint the tile after next into cache (global_prefetch_b8).
    if (kk + 2 * BK < K) {
      if (tid < 64) {
        __builtin_prefetch(Ab + (long long)tid * lda + kk + 2 * BK, 0, 1);
      } else if (BT) {
        __builtin_prefetch(Bb + (long long)(tid - 64) * ldb + kk + 2 * BK, 0, 1);
      } else if (tid - 64 < 32) {
        __builtin_prefetch(Bb + (long long)(kk + 2 * BK + tid - 64) * ldb, 0, 1);
      }
    }
    __syncthreads();   // lds[buf] ready (compiler emits s_wait_dscnt before signal)

    // Fragment loads per documented 16-bit WMMA layouts.
    // A (16x32): lane L -> row = L&15; elems 0..7 = K khi..khi+7, 8..15 = K 16+khi..
    // B (32x16): lane L -> col = L&15; elems 0..7 = K kb..kb+7, 8..15 = K 16+kb.. (kb = 0 or 8)
    const int fr  = lane & 15;
    const int khi = (lane >> 4) << 3;   // 0 or 8
    const int kb  = (lane >> 4) << 3;   // 0 or 8
    v16bf afr[2], bfr[2];
#pragma unroll
    for (int i = 0; i < 2; ++i) {
      v8bf lo = *(const v8bf*)&sA[buf][wm + 16 * i + fr][khi];
      v8bf hi = *(const v8bf*)&sA[buf][wm + 16 * i + fr][16 + khi];
      afr[i] = cat16(lo, hi);
    }
#pragma unroll
    for (int j = 0; j < 2; ++j) {
      v8bf lo = *(const v8bf*)&sB[buf][wn + 16 * j + fr][kb];
      v8bf hi = *(const v8bf*)&sB[buf][wn + 16 * j + fr][kb + 16];
      bfr[j] = cat16(lo, hi);
    }
#pragma unroll
    for (int i = 0; i < 2; ++i)
#pragma unroll
      for (int j = 0; j < 2; ++j)
        acc[i][j] = __builtin_amdgcn_wmma_f32_16x16x32_bf16(
            false, afr[i], false, bfr[j], (short)0, acc[i][j], false, false);

    // Fill the other buffer for the next iteration (no barrier needed here:
    // its previous readers finished before the barrier above).
    if (more) store_tiles(buf ^ 1);
    buf ^= 1;
  }

  // Epilogue. C/D layout: lane L -> n = L&15 ; VGPR r -> m = r + 8*(L>>4)
  const int cn    = lane & 15;
  const int rbase = (lane >> 4) << 3;
#pragma unroll
  for (int i = 0; i < 2; ++i) {
#pragma unroll
    for (int j = 0; j < 2; ++j) {
      const int col  = n0 + wn + 16 * j + cn;
      const int row0 = m0 + wm + 16 * i + rbase;
      float* Cp = Cb + (long long)row0 * ldc + col;
      const float* Rp = Rb ? (Rb + (long long)row0 * ldc + col) : nullptr;
#pragma unroll
      for (int r = 0; r < 8; ++r) {
        float v = acc[i][j][r] * alpha;
        if (act == 1) v = 0.5f * v * (1.0f + erff(v * 0.70710678118654752f)); // exact GELU
        if (Rp) v += Rp[(long long)r * ldc];
        Cp[(long long)r * ldc] = v;
      }
    }
  }
}

// ---------------- elementwise / reduction kernels ----------------
__global__ void embed_kernel(const int* __restrict__ tok,
                             const float* __restrict__ We,
                             const float* __restrict__ Wp,
                             float* __restrict__ x, int total)
{
  int i = blockIdx.x * blockDim.x + threadIdx.x;
  if (i >= total) return;
  int d   = i & (Dc - 1);
  int row = i >> 9;              // token row (Dc = 512)
  int s   = row & (Sc - 1);
  x[i] = We[(long long)tok[row] * Dc + d] + Wp[(long long)s * Dc + d];
}

__global__ __launch_bounds__(128)
void ln_kernel(const float* __restrict__ x, const float* __restrict__ g,
               float* __restrict__ y)
{
  __shared__ float red[128];
  const int row = blockIdx.x;
  const int tid = threadIdx.x;
  const float* xr = x + (long long)row * Dc;
  float v[4]; float s = 0.f;
#pragma unroll
  for (int i = 0; i < 4; ++i) { v[i] = xr[tid + i * 128]; s += v[i]; }
  red[tid] = s; __syncthreads();
  for (int o = 64; o > 0; o >>= 1) { if (tid < o) red[tid] += red[tid + o]; __syncthreads(); }
  const float mu = red[0] * (1.0f / Dc);
  __syncthreads();
  float q = 0.f;
#pragma unroll
  for (int i = 0; i < 4; ++i) { float d = v[i] - mu; q += d * d; }
  red[tid] = q; __syncthreads();
  for (int o = 64; o > 0; o >>= 1) { if (tid < o) red[tid] += red[tid + o]; __syncthreads(); }
  const float inv = rsqrtf(red[0] * (1.0f / Dc) + 1e-5f);
  float* yr = y + (long long)row * Dc;
#pragma unroll
  for (int i = 0; i < 4; ++i) {
    int c = tid + i * 128;
    yr[c] = (v[i] - mu) * inv * g[c];
  }
}

__global__ __launch_bounds__(256)
void softmax_kernel(float* __restrict__ p)   // rows of width S=1024, in place
{
  __shared__ float red[256];
  const int tid = threadIdx.x;
  float* pr = p + (long long)blockIdx.x * Sc;
  float v[4]; float m = -3.4e38f;
#pragma unroll
  for (int i = 0; i < 4; ++i) { v[i] = pr[tid + i * 256]; m = fmaxf(m, v[i]); }
  red[tid] = m; __syncthreads();
  for (int o = 128; o > 0; o >>= 1) { if (tid < o) red[tid] = fmaxf(red[tid], red[tid + o]); __syncthreads(); }
  const float mx = red[0]; __syncthreads();
  float s = 0.f;
#pragma unroll
  for (int i = 0; i < 4; ++i) { v[i] = __expf(v[i] - mx); s += v[i]; }
  red[tid] = s; __syncthreads();
  for (int o = 128; o > 0; o >>= 1) { if (tid < o) red[tid] += red[tid + o]; __syncthreads(); }
  const float inv = 1.0f / red[0];
#pragma unroll
  for (int i = 0; i < 4; ++i) pr[tid + i * 256] = v[i] * inv;
}

__global__ __launch_bounds__(256)
void loss_row_kernel(const float* __restrict__ logits,
                     const int* __restrict__ targets,
                     float* __restrict__ nll)
{
  __shared__ float red[256];
  const int row = blockIdx.x;
  const int tid = threadIdx.x;
  const float* lr = logits + (long long)row * Vc;
  float m = -3.4e38f;
  for (int c = tid; c < Vc; c += 256) m = fmaxf(m, lr[c]);
  red[tid] = m; __syncthreads();
  for (int o = 128; o > 0; o >>= 1) { if (tid < o) red[tid] = fmaxf(red[tid], red[tid + o]); __syncthreads(); }
  const float mx = red[0]; __syncthreads();
  float s = 0.f;
  for (int c = tid; c < Vc; c += 256) s += expf(lr[c] - mx);
  red[tid] = s; __syncthreads();
  for (int o = 128; o > 0; o >>= 1) { if (tid < o) red[tid] += red[tid + o]; __syncthreads(); }
  if (tid == 0) nll[row] = (mx + logf(red[0])) - lr[targets[row]];
}

__global__ __launch_bounds__(256)
void loss_reduce_kernel(const float* __restrict__ nll, float* __restrict__ out)
{
  __shared__ float red[256];
  const int tid = threadIdx.x;
  float s = 0.f;
  for (int i = tid; i < BSc; i += 256) s += nll[i];   // fixed order -> deterministic
  red[tid] = s; __syncthreads();
  for (int o = 128; o > 0; o >>= 1) { if (tid < o) red[tid] += red[tid + o]; __syncthreads(); }
  if (tid == 0) *out = red[0] * (1.0f / BSc);
}

// ---------------- host-side orchestration ----------------
extern "C" void kernel_launch(void* const* d_in, const int* in_sizes, int n_in,
                              void* d_out, int out_size, void* d_ws, size_t ws_size,
                              hipStream_t stream)
{
  const int*   tokens  = (const int*)  d_in[0];
  const int*   targets = (const int*)  d_in[1];
  const float* W_e     = (const float*)d_in[2];
  const float* W_p     = (const float*)d_in[3];
  const float* ln_attn = (const float*)d_in[4];
  const float* Wq      = (const float*)d_in[5];
  const float* Wk      = (const float*)d_in[6];
  const float* Wv      = (const float*)d_in[7];
  const float* Wo      = (const float*)d_in[8];
  const float* ln_ff   = (const float*)d_in[9];
  const float* W_ff1   = (const float*)d_in[10];
  const float* W_ff2   = (const float*)d_in[11];
  const float* ln_out  = (const float*)d_in[12];

  float* logits = (float*)d_out;
  float* loss   = logits + (size_t)BSc * Vc;

  // workspace layout
  char* wsb = (char*)d_ws;
  size_t off = 0;
  auto alloc = [&](size_t bytes) -> float* {
    float* p = (float*)(wsb + off);
    off += (bytes + 255) & ~(size_t)255;
    return p;
  };
  float* x   = alloc((size_t)BSc * Dc * 4);           // residual stream
  float* h   = alloc((size_t)BSc * Dc * 4);           // LN output
  float* Q   = alloc((size_t)Hc * BSc * Dc * 4);      // [H][B*S][D]
  float* Kb  = alloc((size_t)Hc * BSc * Dc * 4);
  float* Vb  = alloc((size_t)Hc * BSc * Dc * 4);
  float* sc  = alloc((size_t)Hc * Bc * Sc * Sc * 4);  // [H][B][S][S]
  float* nll = alloc((size_t)BSc * 4);
  float* o   = Q;    // alias: Q dead after scores
  float* ff  = Kb;   // alias: K dead after scores (needs 32MB <= 64MB)

  const long long SD  = (long long)Sc * Dc;
  const long long BSD = (long long)Bc * Sc * Dc;
  const long long SS  = (long long)Sc * Sc;
  const long long BSS = (long long)Bc * Sc * Sc;
  const float iscale = 0.04419417382415922f; // 1/sqrt(512)

  embed_kernel<<<(BSc * Dc + 255) / 256, 256, 0, stream>>>(tokens, W_e, W_p, x, BSc * Dc);

  for (int l = 0; l < Lc; ++l) {
    const float* Wq_l  = Wq    + (size_t)l * Hc * Dc * Dc;
    const float* Wk_l  = Wk    + (size_t)l * Hc * Dc * Dc;
    const float* Wv_l  = Wv    + (size_t)l * Hc * Dc * Dc;
    const float* Wo_l  = Wo    + (size_t)l * HDc * Dc;
    const float* Wf1_l = W_ff1 + (size_t)l * Dc * DFFc;
    const float* Wf2_l = W_ff2 + (size_t)l * DFFc * Dc;

    ln_kernel<<<BSc, 128, 0, stream>>>(x, ln_attn + l * Dc, h);

    // Q/K/V projections: batched over heads
    {
      dim3 g(Dc / 64, BSc / 64, Hc);
      gemm_wmma<false><<<g, 128, 0, stream>>>(h, Wq_l, nullptr, Q,
          BSc, Dc, Dc, Dc, Dc, Dc,
          Hc, 0, 0, (long long)Dc * Dc, 0, BSD, 0, 1.0f, 0);
      gemm_wmma<false><<<g, 128, 0, stream>>>(h, Wk_l, nullptr, Kb,
          BSc, Dc, Dc, Dc, Dc, Dc,
          Hc, 0, 0, (long long)Dc * Dc, 0, BSD, 0, 1.0f, 0);
      gemm_wmma<false><<<g, 128, 0, stream>>>(h, Wv_l, nullptr, Vb,
          BSc, Dc, Dc, Dc, Dc, Dc,
          Hc, 0, 0, (long long)Dc * Dc, 0, BSD, 0, 1.0f, 0);
    }
    // scores = (Q K^T) * 1/sqrt(D), batched over (b,h): z0=b, z1=h
    {
      dim3 g(Sc / 64, Sc / 64, Bc * Hc);
      gemm_wmma<true><<<g, 128, 0, stream>>>(Q, Kb, nullptr, sc,
          Sc, Sc, Dc, Dc, Dc, Sc,
          Bc, SD, BSD, SD, BSD, SS, BSS, iscale, 0);
    }
    softmax_kernel<<<Hc * Bc * Sc, 256, 0, stream>>>(sc);
    // o = P @ V  -> [B][S][H*D] with head column offset
    {
      dim3 g(Dc / 64, Sc / 64, Bc * Hc);
      gemm_wmma<false><<<g, 128, 0, stream>>>(sc, Vb, nullptr, o,
          Sc, Dc, Sc, Sc, Dc, HDc,
          Bc, SS, BSS, SD, BSD, (long long)Sc * HDc, (long long)Dc, 1.0f, 0);
    }
    // x = x + o @ Wo
    {
      dim3 g(Dc / 64, BSc / 64, 1);
      gemm_wmma<false><<<g, 128, 0, stream>>>(o, Wo_l, x, x,
          BSc, Dc, HDc, HDc, Dc, Dc,
          1, 0, 0, 0, 0, 0, 0, 1.0f, 0);
    }
    ln_kernel<<<BSc, 128, 0, stream>>>(x, ln_ff + l * Dc, h);
    // ff = gelu(h @ W_ff1)
    {
      dim3 g(DFFc / 64, BSc / 64, 1);
      gemm_wmma<false><<<g, 128, 0, stream>>>(h, Wf1_l, nullptr, ff,
          BSc, DFFc, Dc, Dc, DFFc, DFFc,
          1, 0, 0, 0, 0, 0, 0, 1.0f, 1);
    }
    // x = x + ff @ W_ff2
    {
      dim3 g(Dc / 64, BSc / 64, 1);
      gemm_wmma<false><<<g, 128, 0, stream>>>(ff, Wf2_l, x, x,
          BSc, Dc, DFFc, DFFc, Dc, Dc,
          1, 0, 0, 0, 0, 0, 0, 1.0f, 0);
    }
  }

  ln_kernel<<<BSc, 128, 0, stream>>>(x, ln_out, h);
  // logits = h @ W_e^T
  {
    dim3 g(Vc / 64, BSc / 64, 1);
    gemm_wmma<true><<<g, 128, 0, stream>>>(h, W_e, nullptr, logits,
        BSc, Vc, Dc, Dc, Dc, Vc,
        1, 0, 0, 0, 0, 0, 0, 1.0f, 0);
  }
  loss_row_kernel<<<BSc, 256, 0, stream>>>(logits, targets, nll);
  loss_reduce_kernel<<<1, 256, 0, stream>>>(nll, loss);
}
